// NMSLoss_50371376447674
// MI455X (gfx1250) — compile-verified, matrix operand order checked by reading, not run on hardware
//
#include <hip/hip_runtime.h>

#define NMS_THR   0.5f
#define EPS_F     1e-6f
#define NEGK      (-1.0e30f)
#define ALIVE_THR (-1.0e29f)
#define MAXN      2048
#define MAXG      64
#define T         1024
#define NWAVE     (T / 32)

typedef int v8i __attribute__((ext_vector_type(8)));
typedef int v4i __attribute__((ext_vector_type(4)));
typedef unsigned int v4u __attribute__((ext_vector_type(4)));

// ---------------------------------------------------------------------------
// VALU-only wave32 xor-butterfly lane exchange: DPP ROW_XMASK for strides
// 1/2/4/8 inside a 16-lane row, v_permlanex16 (identity selects) to cross the
// halves. No DS pipe, no s_wait_dscnt on the scan's critical path.
// ---------------------------------------------------------------------------
template <int MASK>
__device__ __forceinline__ int lanexor_i(int v) {
  if constexpr (MASK == 16)
    return __builtin_amdgcn_permlanex16(v, v, 0x76543210, 0xFEDCBA98, false, false);
  else
    return __builtin_amdgcn_update_dpp(0, v, 0x160 | MASK, 0xF, 0xF, true);
}
template <int MASK>
__device__ __forceinline__ float lanexor_f(float v) {
  return __int_as_float(lanexor_i<MASK>(__float_as_int(v)));
}

__device__ __forceinline__ float wave_sum_f(float v) {
  v += lanexor_f<1>(v); v += lanexor_f<2>(v); v += lanexor_f<4>(v);
  v += lanexor_f<8>(v); v += lanexor_f<16>(v);
  return v;
}
__device__ __forceinline__ int wave_sum_i(int v) {
  v += lanexor_i<1>(v); v += lanexor_i<2>(v); v += lanexor_i<4>(v);
  v += lanexor_i<8>(v); v += lanexor_i<16>(v);
  return v;
}
__device__ __forceinline__ int wave_or_i(int v) {
  v |= lanexor_i<1>(v); v |= lanexor_i<2>(v); v |= lanexor_i<4>(v);
  v |= lanexor_i<8>(v); v |= lanexor_i<16>(v);
  return v;
}
template <int MASK>
__device__ __forceinline__ void amax_step(float& bv, int& bi) {
  float ov = lanexor_f<MASK>(bv);
  int oi = lanexor_i<MASK>(bi);
  if (ov > bv || (ov == bv && (unsigned)oi < (unsigned)bi)) { bv = ov; bi = oi; }
}
__device__ __forceinline__ void wave_amax(float& bv, int& bi) {
  amax_step<1>(bv, bi); amax_step<2>(bv, bi); amax_step<4>(bv, bi);
  amax_step<8>(bv, bi); amax_step<16>(bv, bi);
}

// ---------------------------------------------------------------------------
// Exact wave32 sum of per-lane counts via V_WMMA_I32_16X16X64_IU8 (ones-B):
// A[L%16][K] holds lane L's count, D rowsums give cnt[m]+cnt[m+16]; each lane
// sums its 8 D rows (= half-total), one xor-16 completes it. Integer -> exact.
// ---------------------------------------------------------------------------
__device__ __forceinline__ int wave_cnt_wmma(int cnt) {
  v8i a = {};
  a[0] = cnt & 0xFF;
  v8i b;
  b[0] = 0x01010101; b[1] = 0x01010101; b[2] = 0x01010101; b[3] = 0x01010101;
  b[4] = 0x01010101; b[5] = 0x01010101; b[6] = 0x01010101; b[7] = 0x01010101;
  v8i c = {};
  v8i d = __builtin_amdgcn_wmma_i32_16x16x64_iu8(false, a, false, b, c, false, false);
  int s = d[0] + d[1] + d[2] + d[3] + d[4] + d[5] + d[6] + d[7];
  s += lanexor_i<16>(s);
  return s;
}

// ---------------------------------------------------------------------------
// Tensor Data Mover: 1-D contiguous DMA of 4-byte elements into LDS.
// D# per CDNA5 ISA: group0 = {count=1 | lds_addr |
// global_addr[56:0] | type=2}; group1 = {data_size=4B, tensor_dim0=tile_dim0=
// nElem, stride0=nElem}; groups 2/3 zero (<=2D tensor).
// ---------------------------------------------------------------------------
__device__ __forceinline__ void tdm_load_1d(unsigned lds_addr, const void* gsrc,
                                            unsigned nElem) {
  unsigned long long ga = (unsigned long long)(uintptr_t)gsrc;
  v4u g0;
  g0[0] = 1u;                                                 // count=1, user D#
  g0[1] = lds_addr;                                           // LDS dest (bytes)
  g0[2] = (unsigned)ga;                                       // global_addr lo
  g0[3] = ((unsigned)(ga >> 32) & 0x01FFFFFFu) | (2u << 30);  // addr[56:32]|type=2
  v8i g1;
  g1[0] = (int)(2u << 16);                         // wg_mask=0, data_size=4B
  g1[1] = (int)((nElem & 0xFFFFu) << 16);          // tensor_dim0[15:0]
  g1[2] = (int)((nElem >> 16) & 0xFFFFu);          // tensor_dim0[31:16], dim1=0
  g1[3] = (int)((nElem & 0xFFFFu) << 16);          // tile_dim0 = nElem
  g1[4] = 0;                                       // tile_dim1/2 unused
  g1[5] = (int)nElem;                              // tensor_dim0_stride lo
  g1[6] = 0;
  g1[7] = 0;
  v4i z4 = {};
#if __clang_major__ >= 23
  v8i z8 = {};
  __builtin_amdgcn_tensor_load_to_lds(g0, g1, z4, z4, z8, 0);
#else
  __builtin_amdgcn_tensor_load_to_lds(g0, g1, z4, z4, 0);
#endif
}

__device__ __forceinline__ float iou_f(float ax1, float ay1, float ax2, float ay2, float aA,
                                       float bx1, float by1, float bx2, float by2, float bA) {
  float iw = fminf(ax2, bx2) - fmaxf(ax1, bx1) + 1.0f;
  float ih = fminf(ay2, by2) - fmaxf(ay1, by1) + 1.0f;
  iw = fmaxf(iw, 0.0f);
  ih = fmaxf(ih, 0.0f);
  float ov = iw * ih;
  return ov / (aA + bA - ov);
}

// ---------------------------------------------------------------------------
// One persistent workgroup per image; LDS-resident sequential NMS scan.
// 3 barriers per scan step; butterflies on VALU; early exit when none alive.
// ---------------------------------------------------------------------------
__global__ __launch_bounds__(T)
void nms_loss_kernel(const int* __restrict__ gt_inds,
                     const float* __restrict__ gt_bboxes,
                     const float* __restrict__ props,
                     float* __restrict__ ws, int N, int G) {
  __shared__ float sX1[MAXN], sY1[MAXN], sX2[MAXN], sY2[MAXN];
  __shared__ float sKey[MAXN];   // alive ? score : NEGK  (reference's NEG mask)
  __shared__ int sGt[MAXN];
  __shared__ float sGB[MAXG * 4];
  __shared__ float sGA[MAXG];
  __shared__ float redF[NWAVE], redF2[NWAVE];
  __shared__ int redI[NWAVE], redC[NWAVE], redA[NWAVE];
  __shared__ int winner[2];
  __shared__ float winnerScore;
  __shared__ int posTot;
  __shared__ int sRec[MAXG];

  const int b = blockIdx.x;
  const int tid = threadIdx.x;
  const int lane = tid & 31;
  const int wid = tid >> 5;

  if (N > MAXN || G > MAXG) {
    if (tid == 0) { ws[2 * b + 0] = 0.0f; ws[2 * b + 1] = 0.0f; }
    return;
  }

  const float* pb = props + (size_t)b * N * 5;
  const int* gi = gt_inds + (size_t)b * N;
  const float* gb = gt_bboxes + (size_t)b * G * 4;

  // Warm caches for the proposal block (global_prefetch_b8).
  for (int off = tid * 64; off < N * 20; off += T * 64)
    __builtin_prefetch((const char*)pb + off, 0, 1);

  if (tid == 0) posTot = 0;

  // TDM DMAs the contiguous gt data into LDS while all waves stage proposals.
  if (wid == 0) {
    tdm_load_1d((unsigned)(uintptr_t)(void*)sGB, gb, (unsigned)(G * 4));
    tdm_load_1d((unsigned)(uintptr_t)(void*)sGt, gi, (unsigned)N);
  }

  for (int j = tid; j < N; j += T) {
    float x1 = pb[j * 5 + 0], y1 = pb[j * 5 + 1];
    float x2 = pb[j * 5 + 2], y2 = pb[j * 5 + 3];
    float sc = pb[j * 5 + 4];
    sX1[j] = x1; sY1[j] = y1; sX2[j] = x2; sY2[j] = y2;
    sKey[j] = sc;
  }
  if (wid == 0) __builtin_amdgcn_s_wait_tensorcnt(0);
  __syncthreads();

  int localPos = 0;
  for (int j = tid; j < N; j += T) {
    int g = sGt[j];
    int p = (g >= 0) ? 1 : 0;
    if (!p) sKey[j] = NEGK;
    localPos += p;
  }
  atomicAdd(&posTot, localPos);
  if (tid < G) {
    sGA[tid] = (sGB[tid * 4 + 2] - sGB[tid * 4 + 0] + 1.0f) *
               (sGB[tid * 4 + 3] - sGB[tid * 4 + 1] + 1.0f);
    sRec[tid] = -1;
  }
  __syncthreads();

  float tpull = 0.0f, tpush = 0.0f;
  int pcnt = 0, qcnt = 0;

  for (int step = 0; step < N; ++step) {
    // ---- argmax over score keys (first-index tie-break, like jnp.argmax) ----
    float bv = -3.0e38f;
    int bi = -1;
    for (int j = tid; j < N; j += T) {
      float s = sKey[j];
      if (s > bv) { bv = s; bi = j; }
    }
    wave_amax(bv, bi);
    if (lane == 0) { redF[wid] = bv; redI[wid] = bi; }
    __syncthreads();  // A1
    if (wid == 0) {
      float v = redF[lane];
      int ix = redI[lane];
      wave_amax(v, ix);
      if (lane == 0) {
        if (v <= ALIVE_THR) {
          winner[0] = -1;  // nothing alive -> remaining scan steps are no-ops
        } else {
          winner[0] = ix;
          int gg = sGt[ix];
          winner[1] = (gg >= 0) ? gg : 0;
          winnerScore = v;
          sKey[ix] = NEGK;  // alive_p = alive \ {i}
        }
      }
    }
    __syncthreads();  // A2
    const int i = winner[0];
    if (i < 0) break;
    const int g = winner[1];

    const float wx1 = sX1[i], wy1 = sY1[i], wx2 = sX2[i], wy2 = sY2[i];
    const float wAr = (wx2 - wx1 + 1.0f) * (wy2 - wy1 + 1.0f);
    const float gx1 = sGB[g * 4 + 0], gy1 = sGB[g * 4 + 1];
    const float gx2 = sGB[g * 4 + 2], gy2 = sGB[g * 4 + 3];
    const float gAr = sGA[g];

    // ---- IoU row, suppression, push-mask terms (single pass, no barrier) ----
    int cnt = 0, anyA = 0;
    float ljsum = 0.0f;
    for (int j = tid; j < N; j += T) {
      float k = sKey[j];
      int a = (k > ALIVE_THR) ? 1 : 0;
      anyA |= a;
      float bx1 = sX1[j], by1 = sY1[j], bx2 = sX2[j], by2 = sY2[j];
      float bAr = (bx2 - bx1 + 1.0f) * (by2 - by1 + 1.0f);
      float r = iou_f(wx1, wy1, wx2, wy2, wAr, bx1, by1, bx2, by2, bAr);
      if (a && r > NMS_THR) {
        sKey[j] = NEGK;  // new_alive = alive_p & ~overlap
        int gj = sGt[j];
        if (gj != g) {
          int sg = (gj >= 0) ? gj : 0;
          float gr = iou_f(gx1, gy1, gx2, gy2, gAr,
                           sGB[sg * 4 + 0], sGB[sg * 4 + 1],
                           sGB[sg * 4 + 2], sGB[sg * 4 + 3], sGA[sg]);
          if (r > gr) { cnt++; ljsum += -logf(1.0f + NMS_THR - r) * k; }
        }
      }
    }

    int wcnt = wave_cnt_wmma(cnt);  // exact count on the matrix pipe
    float wlj = wave_sum_f(ljsum);
    int wany = wave_or_i(anyA);
    if (lane == 0) { redC[wid] = wcnt; redF2[wid] = wlj; redA[wid] = wany; }
    __syncthreads();  // D
    if (wid == 0) {
      int cT = wave_sum_i(redC[lane]);
      float lT = wave_sum_f(redF2[lane]);
      int aT = wave_or_i(redA[lane]);
      if (lane == 0) {
        bool remaining = (aT != 0);
        bool has_rec = (sRec[g] >= 0);
        if (remaining) {
          if (has_rec) {
            int ri = sRec[g];
            float rx1 = sX1[ri], ry1 = sY1[ri], rx2 = sX2[ri], ry2 = sY2[ri];
            float rAr = (rx2 - rx1 + 1.0f) * (ry2 - ry1 + 1.0f);
            float msi = fmaxf(iou_f(wx1, wy1, wx2, wy2, wAr,
                                    rx1, ry1, rx2, ry2, rAr), EPS_F);
            tpull += -logf(msi) * winnerScore;
          }
          tpush += (cT > 0) ? (lT / (float)cT) : 0.0f;
          qcnt += cT;
        }
        if (has_rec) pcnt++; else sRec[g] = i;
      }
    }
    // No trailing barrier needed: next step only writes redF/redI before A1,
    // disjoint from redC/redF2/redA read here; sKey writes happen after A1.
  }

  if (tid == 0) {
    float valid = (posTot > 1) ? 1.0f : 0.0f;
    ws[2 * b + 0] = tpush / ((float)qcnt + EPS_F) * valid;  // push_loss
    ws[2 * b + 1] = tpull / ((float)pcnt + EPS_F) * valid;  // pull_loss
  }
}

__global__ void finalize_kernel(const float* __restrict__ ws, float* __restrict__ out, int B) {
  if (threadIdx.x == 0 && blockIdx.x == 0) {
    float ps = 0.0f, pl = 0.0f;
    for (int b = 0; b < B; ++b) { ps += ws[2 * b + 0]; pl += ws[2 * b + 1]; }
    float invB = 1.0f / (float)B;
    out[0] = ps * invB * 1.0f;  // PUSH_W
    out[1] = pl * invB * 1.0f;  // PULL_W
  }
}

extern "C" void kernel_launch(void* const* d_in, const int* in_sizes, int n_in,
                              void* d_out, int out_size, void* d_ws, size_t ws_size,
                              hipStream_t stream) {
  // inputs: gt_inds (B,N) i32 | anchor_gt_inds (B,N) i32 | gt_bboxes (B,G,4) f32
  //         proposal_list (B,N,5) f32
  const int B = 4;
  const int N = in_sizes[1] / B;
  const int G = in_sizes[2] / (B * 4);
  const int* gt_inds = (const int*)d_in[1];  // reference vmap uses anchor_gt_inds
  const float* gt_bboxes = (const float*)d_in[2];
  const float* props = (const float*)d_in[3];
  float* ws = (float*)d_ws;

  nms_loss_kernel<<<B, T, 0, stream>>>(gt_inds, gt_bboxes, props, ws, N, G);
  finalize_kernel<<<1, 32, 0, stream>>>(ws, (float*)d_out, B);
}
